// HadamardMLPPredictor_73418170957966
// MI455X (gfx1250) — matrix operand order repaired and checked by reading, not compiled, not run on hardware
//
#include <hip/hip_runtime.h>
#include <hip/hip_bf16.h>

// ---------------------------------------------------------------------------
// HadamardMLPPredictor for MI455X (gfx1250), wave32 + WMMA.
//
//   y[e] = relu( BN( (out[src[e]] * out[dst[e]]) @ W1 + b1 ) ) @ W2 + b2
//
// Memory-bound on the 2x512B/edge random gather (~1 GB); GEMM1 (65.5 GFLOP)
// runs on v_wmma_f32_16x16x32_f16. BN needs global batch stats, so we do:
//   k1: fused gather+hadamard+GEMM1 (WMMA), per-block channel sum/sum^2
//       partials to workspace (deterministic, no float atomics).
//   k2: reduce partials -> folded BN coefficients  h' = h*A + B.
//   k3: recompute GEMM1 (cheaper than spilling h to HBM), apply BN+ReLU,
//       fold W2 dot via channel-per-lane layout + 16-lane shfl butterfly.
// Workspace: NB1*512 floats partials (~16 MB) + 512 floats BN coeffs.
// ---------------------------------------------------------------------------

#define E_EDGES   1000000
#define IN_DIM    128
#define HID_DIM   256
#define PADK      136                         // 128 halves + 8 pad (bank spread)
#define EPB       128                         // edges per block (8 waves x 16)
#define NB1       ((E_EDGES + EPB - 1) / EPB) // 7813
#define BN_EPS    1e-5f

typedef __attribute__((ext_vector_type(16))) _Float16 v16h;
typedef __attribute__((ext_vector_type(8)))  _Float16 v8h;
typedef __attribute__((ext_vector_type(8)))  float    v8f;
typedef __attribute__((ext_vector_type(4)))  float    v4f;

// Build the f16 A-fragment (16x32 tile, ISA layout) with the Hadamard product
// fused into the f32->f16 conversion.
// Lane L (row = L&15):  halves 0..7  = K kb..kb+7,  halves 8..15 = K kb+16..kb+23,
// where kb = c*32 + (L>>4)*8.
__device__ __forceinline__ v16h make_a_frag(const float* __restrict__ ps,
                                            const float* __restrict__ pd,
                                            int kb) {
    const v4f* s = (const v4f*)(ps + kb);
    const v4f* d = (const v4f*)(pd + kb);
    v4f s0 = s[0], s1 = s[1], s2 = s[4], s3 = s[5];
    v4f d0 = d[0], d1 = d[1], d2 = d[4], d3 = d[5];
    v16h a;
#pragma unroll
    for (int i = 0; i < 4; ++i) {
        a[i]      = (_Float16)(s0[i] * d0[i]);
        a[4 + i]  = (_Float16)(s1[i] * d1[i]);
        a[8 + i]  = (_Float16)(s2[i] * d2[i]);
        a[12 + i] = (_Float16)(s3[i] * d3[i]);
    }
    return a;
}

// B-fragment (32x16 tile of W1) from LDS column-major f16.
// Lane L: column N = n*16 + (L&15); halves 0..7 = K kb..kb+7, halves 8..15 = K kb+16..kb+23, kb = c*32 + (L>>4)*8.
__device__ __forceinline__ v16h load_b_frag(const _Float16* __restrict__ sW1,
                                            int n, int c, int lm, int half) {
    const _Float16* p = sW1 + (n * 16 + lm) * PADK + c * 32 + half * 8;
    v8h lo = *(const v8h*)p;
    v8h hi = *(const v8h*)(p + 16);
    return __builtin_shufflevector(lo, hi, 0, 1, 2, 3, 4, 5, 6, 7,
                                   8, 9, 10, 11, 12, 13, 14, 15);
}

__device__ __forceinline__ void load_w1_lds(_Float16* sW1,
                                            const float* __restrict__ W1,
                                            int tid) {
    // W1 is row-major [IN_DIM][HID_DIM]; store column-major f16, padded.
    for (int i = tid; i < IN_DIM * HID_DIM; i += 256) {
        int k = i >> 8;        // row (K)
        int n = i & 255;       // col (channel)
        sW1[n * PADK + k] = (_Float16)W1[i];
    }
}

// ---------------- kernel 1: GEMM1 + per-block channel stats ----------------
__global__ __launch_bounds__(256)
void k1_gemm_stats(const float* __restrict__ nodes,
                   const int*   __restrict__ eidx,   // [2][E]
                   const float* __restrict__ W1,
                   const float* __restrict__ b1,
                   float*       __restrict__ partials /* [NB1][512] */) {
    __shared__ _Float16 sW1[HID_DIM * PADK];
    __shared__ float sSum[8][HID_DIM];
    __shared__ float sSq [8][HID_DIM];
    __shared__ float sB1 [HID_DIM];

    const int tid = threadIdx.x;
    load_w1_lds(sW1, W1, tid);
    if (tid < HID_DIM) sB1[tid] = b1[tid];
    __syncthreads();

    const int wave = tid >> 5, lane = tid & 31;
    const int half = lane >> 4, lm = lane & 15;
    const int e0 = blockIdx.x * EPB + wave * 16;

    int er = e0 + lm;
    int eg = er < E_EDGES ? er : (E_EDGES - 1);   // clamp gather for tail
    const int si = eidx[eg];
    const int di = eidx[(size_t)E_EDGES + eg];
    const float* ps = nodes + (size_t)si * IN_DIM;
    const float* pd = nodes + (size_t)di * IN_DIM;

    v16h a[4];
#pragma unroll
    for (int c = 0; c < 4; ++c) a[c] = make_a_frag(ps, pd, c * 32 + half * 8);

#pragma unroll
    for (int n = 0; n < 16; ++n) {
        const int ch = n * 16 + lm;
        float bias = sB1[ch];
        v8f acc;
#pragma unroll
        for (int j = 0; j < 8; ++j) acc[j] = bias;
#pragma unroll
        for (int c = 0; c < 4; ++c) {
            v16h b = load_b_frag(sW1, n, c, lm, half);
            acc = __builtin_amdgcn_wmma_f32_16x16x32_f16(
                false, a[c], false, b, (short)0, acc, false, false);
        }
        // D layout: lane -> channel ch; VGPR j -> edge e0 + j + half*8.
        float s = 0.f, s2 = 0.f;
#pragma unroll
        for (int j = 0; j < 8; ++j) {
            int e = e0 + j + half * 8;
            if (e < E_EDGES) { float v = acc[j]; s += v; s2 += v * v; }
        }
        // combine the two half-wave copies of this channel
        s  += __shfl_xor(s, 16);
        s2 += __shfl_xor(s2, 16);
        if (half == 0) { sSum[wave][ch] = s; sSq[wave][ch] = s2; }
    }
    __syncthreads();

    // deterministic per-block reduction across the 8 waves
    float ts = 0.f, ts2 = 0.f;
#pragma unroll
    for (int w = 0; w < 8; ++w) { ts += sSum[w][tid]; ts2 += sSq[w][tid]; }
    float* pb = partials + (size_t)blockIdx.x * (2 * HID_DIM);
    pb[tid]           = ts;
    pb[HID_DIM + tid] = ts2;
}

// ---------------- kernel 2: reduce partials -> folded BN coeffs ------------
__global__ __launch_bounds__(256)
void k2_finalize(const float* __restrict__ partials,
                 const float* __restrict__ gamma,
                 const float* __restrict__ beta,
                 float*       __restrict__ bn /* [2][256]: A then B */) {
    const int c = blockIdx.x;     // channel 0..255
    const int t = threadIdx.x;
    float s = 0.f, s2 = 0.f;
    for (int b = t; b < NB1; b += 256) {
        const float* pb = partials + (size_t)b * (2 * HID_DIM);
        s  += pb[c];
        s2 += pb[HID_DIM + c];
    }
    __shared__ float rs[256], rq[256];
    rs[t] = s; rq[t] = s2;
    __syncthreads();
#pragma unroll
    for (int off = 128; off > 0; off >>= 1) {
        if (t < off) { rs[t] += rs[t + off]; rq[t] += rq[t + off]; }
        __syncthreads();
    }
    if (t == 0) {
        const float invE = 1.0f / (float)E_EDGES;
        float mean = rs[0] * invE;
        float var  = rq[0] * invE - mean * mean;
        float inv  = rsqrtf(var + BN_EPS);
        float A = gamma[c] * inv;
        bn[c]           = A;
        bn[HID_DIM + c] = beta[c] - mean * A;
    }
}

// ---------------- kernel 3: recompute GEMM1, BN+ReLU, dot W2, write y ------
__global__ __launch_bounds__(256)
void k3_output(const float* __restrict__ nodes,
               const int*   __restrict__ eidx,
               const float* __restrict__ W1,
               const float* __restrict__ b1,
               const float* __restrict__ bn,
               const float* __restrict__ W2,
               const float* __restrict__ b2,
               float*       __restrict__ y) {
    __shared__ _Float16 sW1[HID_DIM * PADK];
    __shared__ float sA[HID_DIM], sB[HID_DIM], sW2[HID_DIM], sB1[HID_DIM];

    const int tid = threadIdx.x;
    load_w1_lds(sW1, W1, tid);
    if (tid < HID_DIM) {
        sA[tid]  = bn[tid];
        sB[tid]  = bn[HID_DIM + tid];
        sW2[tid] = W2[tid];
        sB1[tid] = b1[tid];
    }
    __syncthreads();

    const int wave = tid >> 5, lane = tid & 31;
    const int half = lane >> 4, lm = lane & 15;
    const int e0 = blockIdx.x * EPB + wave * 16;

    int er = e0 + lm;
    int eg = er < E_EDGES ? er : (E_EDGES - 1);
    const int si = eidx[eg];
    const int di = eidx[(size_t)E_EDGES + eg];
    const float* ps = nodes + (size_t)si * IN_DIM;
    const float* pd = nodes + (size_t)di * IN_DIM;

    v16h a[4];
#pragma unroll
    for (int c = 0; c < 4; ++c) a[c] = make_a_frag(ps, pd, c * 32 + half * 8);

    float yacc[8];
#pragma unroll
    for (int j = 0; j < 8; ++j) yacc[j] = 0.f;

#pragma unroll
    for (int n = 0; n < 16; ++n) {
        const int ch = n * 16 + lm;
        float bias = sB1[ch];
        v8f acc;
#pragma unroll
        for (int j = 0; j < 8; ++j) acc[j] = bias;
#pragma unroll
        for (int c = 0; c < 4; ++c) {
            v16h b = load_b_frag(sW1, n, c, lm, half);
            acc = __builtin_amdgcn_wmma_f32_16x16x32_f16(
                false, a[c], false, b, (short)0, acc, false, false);
        }
        float A = sA[ch], Bv = sB[ch], w2v = sW2[ch];
#pragma unroll
        for (int j = 0; j < 8; ++j) {
            float h = fmaf(acc[j], A, Bv);      // folded BN
            h = fmaxf(h, 0.f);                  // ReLU
            yacc[j] = fmaf(h, w2v, yacc[j]);    // * W2[ch]
        }
    }

    // reduce across the 16 lanes of each half (channels) -> per-edge scalar
#pragma unroll
    for (int j = 0; j < 8; ++j) {
        float v = yacc[j];
        v += __shfl_xor(v, 1);
        v += __shfl_xor(v, 2);
        v += __shfl_xor(v, 4);
        v += __shfl_xor(v, 8);
        yacc[j] = v;
    }
    if (lm == 0) {
        float bb = b2[0];
#pragma unroll
        for (int j = 0; j < 8; ++j) {
            int e = e0 + j + half * 8;
            if (e < E_EDGES) y[e] = yacc[j] + bb;
        }
    }
}

// ---------------------------------------------------------------------------
extern "C" void kernel_launch(void* const* d_in, const int* in_sizes, int n_in,
                              void* d_out, int out_size, void* d_ws, size_t ws_size,
                              hipStream_t stream) {
    const float* nodes = (const float*)d_in[0];   // [500000][128] f32
    const int*   eidx  = (const int*)  d_in[1];   // [2][1000000] int
    const float* W1    = (const float*)d_in[2];   // [128][256]
    const float* b1    = (const float*)d_in[3];   // [256]
    const float* gamma = (const float*)d_in[4];   // [256]
    const float* beta  = (const float*)d_in[5];   // [256]
    const float* W2    = (const float*)d_in[6];   // [256][1]
    const float* b2    = (const float*)d_in[7];   // [1]
    float*       y     = (float*)d_out;           // [1000000]

    float* partials = (float*)d_ws;                       // NB1*512 floats
    float* bn       = partials + (size_t)NB1 * (2 * HID_DIM); // 512 floats

    k1_gemm_stats<<<NB1, 256, 0, stream>>>(nodes, eidx, W1, b1, partials);
    k2_finalize  <<<HID_DIM, 256, 0, stream>>>(partials, gamma, beta, bn);
    k3_output    <<<NB1, 256, 0, stream>>>(nodes, eidx, W1, b1, bn, W2, b2, y);
}
